// TransformerDecoder_87187836108888
// MI455X (gfx1250) — matrix operand order repaired and checked by reading, not compiled, not run on hardware
//
#include <hip/hip_runtime.h>
#include <stdint.h>

#define DEVINL __device__ __forceinline__

// Problem constants (match reference)
constexpr int BB = 16, TT = 500, CC = 512, KK = 7, NHh = 8, HD = 64, HIDd = 2048, LL = 3, NCB = 2;
constexpr int MM = BB * TT;             // 8000 rows (b,t)
constexpr float ASCALE = 0.125f;        // 1/sqrt(HD=64)

typedef __attribute__((ext_vector_type(16))) __bf16 v16bf;
typedef __attribute__((ext_vector_type(8)))  float  v8f;

union AF { v16bf v; uint32_t u[8]; uint4 q[2]; };  // 16x32 bf16 A/B fragment
union CF { v8f   v; float    f[8]; };              // 16x16 f32 C/D fragment

DEVINL uint16_t f2bf(float f) {         // RNE float -> bf16
  uint32_t u = __float_as_uint(f);
  u += 0x7FFFu + ((u >> 16) & 1u);
  return (uint16_t)(u >> 16);
}
DEVINL float bf2f(uint16_t h) { return __uint_as_float(((uint32_t)h) << 16); }

DEVINL v8f wmma_bf16(const AF& a, const AF& b, v8f c) {
  return __builtin_amdgcn_wmma_f32_16x16x32_bf16(false, a.v, false, b.v,
                                                 (short)0, c, false, false);
}

// ---------------------------------------------------------------------------
// fp32 -> bf16 cast (input x once + weight cache once per launch)
// ---------------------------------------------------------------------------
__global__ void cast_f32_bf16_kernel(const float* __restrict__ in,
                                     uint16_t* __restrict__ out, int n) {
  int i = blockIdx.x * 256 + threadIdx.x;
  if (i < n) out[i] = f2bf(in[i]);
}

// ---------------------------------------------------------------------------
// Depthwise conv over T (groups=C, K=7, same padding), bf16 in/out.
// ---------------------------------------------------------------------------
__global__ __launch_bounds__(256) void dwconv_kernel(
    const uint16_t* __restrict__ in, const float* __restrict__ w,
    const float* __restrict__ bias, uint16_t* __restrict__ out) {
  const int idx = blockIdx.x * 256 + threadIdx.x;   // < MM*CC
  const int c = idx & (CC - 1);
  const int m = idx >> 9;                           // CC == 512
  const int t = m % TT;
  const int base = m - t;                           // first row of this batch
  float acc = bias[c];
#pragma unroll
  for (int k = 0; k < KK; ++k) {
    const int tt2 = t + k - KK / 2;
    if (tt2 >= 0 && tt2 < TT)
      acc += bf2f(in[(size_t)(base + tt2) * CC + c]) * w[c * KK + k];
  }
  out[(size_t)m * CC + c] = f2bf(acc);
}

// ---------------------------------------------------------------------------
// WMMA GEMM: Y(M,N) = X(M,Kd) * W(N,Kd)^T + bias, optional relu.
// X bf16, W bf16 (pre-converted cache). Block = 256 thr (8 waves = 2x4),
// tile 64(M) x 256(N); each wave 32x64 -> 8 wmma per 32-k step vs 12 b128 ds.
// ---------------------------------------------------------------------------
template <bool RELU, bool OUTBF>
__global__ __launch_bounds__(256) void gemm_xwt_kernel(
    const uint16_t* __restrict__ X, const uint16_t* __restrict__ Wb,
    const float* __restrict__ bias, void* __restrict__ Yv, int Kd, int N) {
  __shared__ uint16_t As[64][40];     // 64 rows x 32 k (pad 40: 80B stride)
  __shared__ uint16_t Bs[256][40];    // 256 out-cols x 32 k
  const int tid = threadIdx.x;
  const int lane = tid & 31;
  const int wid = tid >> 5;
  const int wm = (wid & 1) * 32;      // wave M offset in tile (2 waves)
  const int wn = (wid >> 1) * 64;     // wave N offset in tile (4 waves)
  const int m0 = blockIdx.x * 64;
  const int n0 = blockIdx.y * 256;
  const int laneN = lane & 15;
  const int laneHi = lane >> 4;

  CF acc[2][4];
#pragma unroll
  for (int mt = 0; mt < 2; ++mt)
#pragma unroll
    for (int i = 0; i < 4; ++i)
#pragma unroll
      for (int j = 0; j < 8; ++j) acc[mt][i].f[j] = 0.0f;

  const int xrow = tid >> 2, xcol = (tid & 3) * 8;   // X tile: 16B per thread

  for (int k0 = 0; k0 < Kd; k0 += 32) {
    // stage X tile (64x32 bf16)
    *(uint4*)&As[xrow][xcol] =
        *(const uint4*)(X + (size_t)(m0 + xrow) * Kd + k0 + xcol);
    // stage W tile (256x32 bf16): one row per thread, 64B
    {
      const uint16_t* srcw = Wb + (size_t)(n0 + tid) * Kd + k0;
      uint4* dst = (uint4*)&Bs[tid][0];
      const uint4* s4 = (const uint4*)srcw;
      dst[0] = s4[0]; dst[1] = s4[1]; dst[2] = s4[2]; dst[3] = s4[3];
    }
    if (k0 + 32 < Kd) {  // global_prefetch_b8 for next k-tile
      __builtin_prefetch(X + (size_t)(m0 + xrow) * Kd + k0 + 32 + xcol, 0, 0);
      __builtin_prefetch(Wb + (size_t)(n0 + tid) * Kd + k0 + 32, 0, 0);
    }
    __syncthreads();

    // A fragments: per ISA layout, VGPRs 0-3 / 4-7 are 16B-contiguous per lane
    AF a[2];
#pragma unroll
    for (int mt = 0; mt < 2; ++mt) {
      const uint16_t* ar = &As[wm + mt * 16 + laneN][laneHi * 8];
      a[mt].q[0] = *(const uint4*)ar;          // k 0..15 half
      a[mt].q[1] = *(const uint4*)(ar + 16);   // k 16..31 half
    }
#pragma unroll
    for (int nt = 0; nt < 4; ++nt) {
      AF bfr;  // B fragment: same k layout as A (16B halves at laneHi*8, 16+laneHi*8)
      const uint16_t* br = &Bs[wn + nt * 16 + laneN][laneHi * 8];
      bfr.q[0] = *(const uint4*)br;
      bfr.q[1] = *(const uint4*)(br + 16);
#pragma unroll
      for (int mt = 0; mt < 2; ++mt)
        acc[mt][nt].v = wmma_bf16(a[mt], bfr, acc[mt][nt].v);
    }
    __syncthreads();
  }

  // epilogue (C layout: VGPR r -> rows r / r+8, lane%16 -> col)
#pragma unroll
  for (int nt = 0; nt < 4; ++nt) {
    const int col = n0 + wn + nt * 16 + laneN;
    const float bvv = bias[col];
#pragma unroll
    for (int mt = 0; mt < 2; ++mt)
#pragma unroll
      for (int r = 0; r < 8; ++r) {
        const int row = m0 + wm + mt * 16 + r + laneHi * 8;
        float val = acc[mt][nt].f[r] + bvv;
        if (RELU) val = fmaxf(val, 0.0f);
        if (OUTBF)
          ((uint16_t*)Yv)[(size_t)row * N + col] = f2bf(val);
        else
          ((float*)Yv)[(size_t)row * N + col] = val;
      }
  }
}

// ---------------------------------------------------------------------------
// Flash attention (causal): one wave per (b*NH, 16-query tile).
// ---------------------------------------------------------------------------
__global__ __launch_bounds__(32) void flash_attn_kernel(
    const uint16_t* __restrict__ Q, const uint16_t* __restrict__ Km,
    const uint16_t* __restrict__ V, uint16_t* __restrict__ O) {
  __shared__ uint16_t pls[16 * 40];   // P re-stripe buffer (pad 40: 80B stride)
  const int lane = threadIdx.x;
  const int laneN = lane & 15, laneHi = lane >> 4;
  const int bh = blockIdx.x;
  const int b = bh >> 3;              // NH == 8
  const int h = bh & 7;
  const int q0 = blockIdx.y * 16;

  // Q fragments (16x64 -> two 16x32 A frags), 2x b128 each
  AF qa[2];
  {
    int row = q0 + laneN; if (row > TT - 1) row = TT - 1;
    const uint16_t* qp = Q + (size_t)(b * TT + row) * CC + h * HD;
#pragma unroll
    for (int dd = 0; dd < 2; ++dd) {
      qa[dd].q[0] = *(const uint4*)(qp + dd * 32 + laneHi * 8);
      qa[dd].q[1] = *(const uint4*)(qp + dd * 32 + 16 + laneHi * 8);
    }
  }

  float mrow[8], lrow[8];
  CF o[4];
#pragma unroll
  for (int r = 0; r < 8; ++r) { mrow[r] = -3.0e38f; lrow[r] = 0.0f; }
#pragma unroll
  for (int dt = 0; dt < 4; ++dt)
#pragma unroll
    for (int r = 0; r < 8; ++r) o[dt].f[r] = 0.0f;

  const uint16_t* vbase = V + (size_t)b * TT * CC + h * HD;
  const int kend = (q0 + 16 < TT) ? (q0 + 16) : TT;   // causal pruning
  for (int k0 = 0; k0 < kend; k0 += 32) {
    CF sc[2];
#pragma unroll
    for (int ct = 0; ct < 2; ++ct) {
#pragma unroll
      for (int r = 0; r < 8; ++r) sc[ct].f[r] = 0.0f;
      int key = k0 + ct * 16 + laneN; if (key > TT - 1) key = TT - 1;
      const uint16_t* kp = Km + (size_t)(b * TT + key) * CC + h * HD;
#pragma unroll
      for (int dd = 0; dd < 2; ++dd) {
        AF kb;  // B frag: same k layout as A -> 2x b128
        kb.q[0] = *(const uint4*)(kp + dd * 32 + laneHi * 8);
        kb.q[1] = *(const uint4*)(kp + dd * 32 + 16 + laneHi * 8);
        sc[ct].v = wmma_bf16(qa[dd], kb, sc[ct].v);
      }
    }
    // scale + causal mask (C layout: row = r + 8*laneHi, col = laneN)
#pragma unroll
    for (int ct = 0; ct < 2; ++ct)
#pragma unroll
      for (int r = 0; r < 8; ++r) {
        const int qi = q0 + r + laneHi * 8;
        const int ki = k0 + ct * 16 + laneN;
        float v = sc[ct].f[r] * ASCALE;
        if (ki > qi || ki >= TT) v = -1.0e30f;
        sc[ct].f[r] = v;
      }
    // online softmax
#pragma unroll
    for (int r = 0; r < 8; ++r) {
      float mx = fmaxf(sc[0].f[r], sc[1].f[r]);
#pragma unroll
      for (int m = 1; m < 16; m <<= 1) mx = fmaxf(mx, __shfl_xor(mx, m, 32));
      const float mn = fmaxf(mrow[r], mx);
      const float alpha = __expf(mrow[r] - mn);
      mrow[r] = mn;
      lrow[r] *= alpha;
#pragma unroll
      for (int dt = 0; dt < 4; ++dt) o[dt].f[r] *= alpha;
      const float p0 = __expf(sc[0].f[r] - mn);
      const float p1 = __expf(sc[1].f[r] - mn);
      sc[0].f[r] = p0; sc[1].f[r] = p1;
      float ps = p0 + p1;
#pragma unroll
      for (int m = 1; m < 16; m <<= 1) ps += __shfl_xor(ps, m, 32);
      lrow[r] += ps;
    }
    // re-stripe P (C layout) -> A layout via LDS
#pragma unroll
    for (int ct = 0; ct < 2; ++ct)
#pragma unroll
      for (int r = 0; r < 8; ++r)
        pls[(r + laneHi * 8) * 40 + ct * 16 + laneN] = f2bf(sc[ct].f[r]);
    __syncthreads();
    AF pa;
    pa.q[0] = *(const uint4*)&pls[laneN * 40 + laneHi * 8];
    pa.q[1] = *(const uint4*)&pls[laneN * 40 + 16 + laneHi * 8];
    __syncthreads();
    // P @ V: hoist the 16 V row pointers (k-pair spans two key rows)
    const uint16_t* vr0[8];
    const uint16_t* vr1[8];
#pragma unroll
    for (int v = 0; v < 8; ++v) {
      int kk0 = k0 + 2 * (v & 3) + (v >> 2) * 16 + laneHi * 8;
      int kk1 = kk0 + 1;
      if (kk0 > TT - 1) kk0 = TT - 1;
      if (kk1 > TT - 1) kk1 = TT - 1;
      vr0[v] = vbase + (size_t)kk0 * CC;
      vr1[v] = vbase + (size_t)kk1 * CC;
    }
#pragma unroll
    for (int dt = 0; dt < 4; ++dt) {
      AF vbf;
      const int d = dt * 16 + laneN;
#pragma unroll
      for (int v = 0; v < 8; ++v)
        vbf.u[v] = (uint32_t)vr0[v][d] | ((uint32_t)vr1[v][d] << 16);
      o[dt].v = wmma_bf16(pa, vbf, o[dt].v);
    }
  }
  // write O (bf16), skip rows >= T
#pragma unroll
  for (int dt = 0; dt < 4; ++dt)
#pragma unroll
    for (int r = 0; r < 8; ++r) {
      const int row = q0 + r + laneHi * 8;
      if (row < TT) {
        const float val = o[dt].f[r] / lrow[r];
        O[(size_t)(b * TT + row) * CC + h * HD + dt * 16 + laneN] = f2bf(val);
      }
    }
}

// ---------------------------------------------------------------------------
// LayerNorm over C=512: one wave per row; writes bf16 and/or fp32.
// ---------------------------------------------------------------------------
__global__ __launch_bounds__(32) void ln_kernel(
    const float* __restrict__ in, const float* __restrict__ g,
    const float* __restrict__ be, uint16_t* __restrict__ obf,
    float* __restrict__ of) {
  const int row = blockIdx.x;
  const int lane = threadIdx.x;
  const float* r = in + (size_t)row * CC;
  float x[16], s = 0.0f, s2 = 0.0f;
#pragma unroll
  for (int i = 0; i < 16; ++i) {
    x[i] = r[i * 32 + lane];
    s += x[i]; s2 += x[i] * x[i];
  }
#pragma unroll
  for (int m = 1; m < 32; m <<= 1) {
    s += __shfl_xor(s, m, 32);
    s2 += __shfl_xor(s2, m, 32);
  }
  const float mean = s * (1.0f / CC);
  const float var = s2 * (1.0f / CC) - mean * mean;
  const float rstd = rsqrtf(var + 1e-5f);
#pragma unroll
  for (int i = 0; i < 16; ++i) {
    const int c = i * 32 + lane;
    const float v = (x[i] - mean) * rstd * g[c] + be[c];
    if (obf) obf[(size_t)row * CC + c] = f2bf(v);
    if (of) of[(size_t)row * CC + c] = v;
  }
}

// ---------------------------------------------------------------------------
// Host orchestration. Workspace layout (~96 MB):
//   a0,a1,a2      : bf16 M*C activations (ping-pong + layer carrier)
//   qkv region    : bf16 M*HID (q,k,v during attention; reused as FFN hidden)
//   tf            : fp32 M*C (pre-LayerNorm GEMM outputs)
//   weight cache  : bf16 copies of all GEMM weights (converted once)
// ---------------------------------------------------------------------------
extern "C" void kernel_launch(void* const* d_in, const int* in_sizes, int n_in,
                              void* d_out, int out_size, void* d_ws, size_t ws_size,
                              hipStream_t stream) {
  (void)in_sizes; (void)n_in; (void)out_size; (void)ws_size;
  const float* x     = (const float*)d_in[0];
  const float* dw_w  = (const float*)d_in[1];
  const float* dw_b  = (const float*)d_in[2];
  const float* pw_w  = (const float*)d_in[3];
  const float* pw_b  = (const float*)d_in[4];
  const float* cln_g = (const float*)d_in[5];
  const float* cln_b = (const float*)d_in[6];
  const float* wq = (const float*)d_in[7];  const float* bq = (const float*)d_in[8];
  const float* wk = (const float*)d_in[9];  const float* bk = (const float*)d_in[10];
  const float* wv = (const float*)d_in[11]; const float* bv = (const float*)d_in[12];
  const float* wo = (const float*)d_in[13]; const float* bo = (const float*)d_in[14];
  const float* aln_g = (const float*)d_in[15]; const float* aln_b = (const float*)d_in[16];
  const float* w1 = (const float*)d_in[17]; const float* b1 = (const float*)d_in[18];
  const float* w2 = (const float*)d_in[19]; const float* b2 = (const float*)d_in[20];
  const float* fln_g = (const float*)d_in[21]; const float* fln_b = (const float*)d_in[22];
  float* out = (float*)d_out;

  const size_t MC = (size_t)MM * CC;
  const size_t WCC = (size_t)CC * CC;
  uint16_t* a0 = (uint16_t*)d_ws;
  uint16_t* a1 = a0 + MC;
  uint16_t* a2 = a1 + MC;
  uint16_t* qkv = a2 + MC;                       // size MM*HID elems
  uint16_t* qb = qkv;
  uint16_t* kb = qkv + MC;
  uint16_t* vb = qkv + 2 * MC;
  uint16_t* hid = qkv;                           // reuse after attention
  float* tf = (float*)(qkv + (size_t)MM * HIDd);
  // bf16 weight cache
  uint16_t* pwbf = (uint16_t*)(tf + MC);         // L*NC * C*C
  uint16_t* wqbf = pwbf + (size_t)LL * NCB * WCC;
  uint16_t* wkbf = wqbf + (size_t)LL * WCC;
  uint16_t* wvbf = wkbf + (size_t)LL * WCC;
  uint16_t* wobf = wvbf + (size_t)LL * WCC;
  uint16_t* w1bf = wobf + (size_t)LL * WCC;      // L * HID*C
  uint16_t* w2bf = w1bf + (size_t)LL * HIDd * CC;

  const dim3 blk256(256), blk32(32);
  const dim3 gEw((MM * CC) / 256);               // elementwise / dwconv grid
  const dim3 gGc(MM / 64, CC / 256);             // GEMM N=512 (tile 64x256)
  const dim3 gGh(MM / 64, HIDd / 256);           // GEMM N=2048
  const dim3 gLn(MM);
  const dim3 gAtt(BB * NHh, (TT + 15) / 16);
  auto cdiv = [](int a, int b) { return (a + b - 1) / b; };

  // one-time conversions: input x and all GEMM weights -> bf16
  cast_f32_bf16_kernel<<<gEw, blk256, 0, stream>>>(x, a2, MM * CC);
  cast_f32_bf16_kernel<<<dim3(cdiv(LL * NCB * (int)WCC, 256)), blk256, 0, stream>>>(
      pw_w, pwbf, LL * NCB * (int)WCC);
  cast_f32_bf16_kernel<<<dim3(cdiv(LL * (int)WCC, 256)), blk256, 0, stream>>>(wq, wqbf, LL * (int)WCC);
  cast_f32_bf16_kernel<<<dim3(cdiv(LL * (int)WCC, 256)), blk256, 0, stream>>>(wk, wkbf, LL * (int)WCC);
  cast_f32_bf16_kernel<<<dim3(cdiv(LL * (int)WCC, 256)), blk256, 0, stream>>>(wv, wvbf, LL * (int)WCC);
  cast_f32_bf16_kernel<<<dim3(cdiv(LL * (int)WCC, 256)), blk256, 0, stream>>>(wo, wobf, LL * (int)WCC);
  cast_f32_bf16_kernel<<<dim3(cdiv(LL * HIDd * CC, 256)), blk256, 0, stream>>>(w1, w1bf, LL * HIDd * CC);
  cast_f32_bf16_kernel<<<dim3(cdiv(LL * HIDd * CC, 256)), blk256, 0, stream>>>(w2, w2bf, LL * HIDd * CC);

  for (int l = 0; l < LL; ++l) {
    const uint16_t* cur = a2;
    for (int cb = 0; cb < NCB; ++cb) {
      const int lc = l * NCB + cb;
      dwconv_kernel<<<gEw, blk256, 0, stream>>>(
          cur, dw_w + (size_t)lc * CC * KK, dw_b + (size_t)lc * CC, a0);
      gemm_xwt_kernel<false, false><<<gGc, blk256, 0, stream>>>(
          a0, pwbf + (size_t)lc * WCC, pw_b + (size_t)lc * CC, tf, CC, CC);
      ln_kernel<<<gLn, blk32, 0, stream>>>(
          tf, cln_g + (size_t)lc * CC, cln_b + (size_t)lc * CC, a1, nullptr);
      cur = a1;
    }
    // attention: QKV projections -> flash -> output projection -> LN
    gemm_xwt_kernel<false, true><<<gGc, blk256, 0, stream>>>(
        a1, wqbf + (size_t)l * WCC, bq + (size_t)l * CC, qb, CC, CC);
    gemm_xwt_kernel<false, true><<<gGc, blk256, 0, stream>>>(
        a1, wkbf + (size_t)l * WCC, bk + (size_t)l * CC, kb, CC, CC);
    gemm_xwt_kernel<false, true><<<gGc, blk256, 0, stream>>>(
        a1, wvbf + (size_t)l * WCC, bv + (size_t)l * CC, vb, CC, CC);
    flash_attn_kernel<<<gAtt, blk32, 0, stream>>>(qb, kb, vb, a0);
    gemm_xwt_kernel<false, false><<<gGc, blk256, 0, stream>>>(
        a0, wobf + (size_t)l * WCC, bo + (size_t)l * CC, tf, CC, CC);
    ln_kernel<<<gLn, blk32, 0, stream>>>(
        tf, aln_g + (size_t)l * CC, aln_b + (size_t)l * CC, a1, nullptr);
    // FFN: relu(X W1^T + b1) W2^T + b2, then final LN -> d_out + carrier
    gemm_xwt_kernel<true, true><<<gGh, blk256, 0, stream>>>(
        a1, w1bf + (size_t)l * HIDd * CC, b1 + (size_t)l * HIDd, hid, CC, HIDd);
    gemm_xwt_kernel<false, false><<<gGc, blk256, 0, stream>>>(
        hid, w2bf + (size_t)l * CC * HIDd, b2 + (size_t)l * CC, tf, HIDd, CC);
    ln_kernel<<<gLn, blk32, 0, stream>>>(
        tf, fln_g + (size_t)l * CC, fln_b + (size_t)l * CC, a2,
        out + (size_t)l * MM * CC);
  }
}
